// InterSentenceInteraction_85847806313068
// MI455X (gfx1250) — matrix operand order repaired and checked by reading, not compiled, not run on hardware
//
#include <hip/hip_runtime.h>
#include <stdint.h>

#define B_  32
#define LA  512
#define LB  512
#define DD  1024

// LDS B-tile: 128 rows x 64 K (bf16), rows padded to 72 elems (144B) for
// conflict-free ds_load_b128 fragment reads. Double buffered.
#define BT_ROWPAD 72
#define BT_ELEMS  (128 * BT_ROWPAD)

typedef __attribute__((ext_vector_type(16))) __bf16 v16bf;
typedef __attribute__((ext_vector_type(8)))  float  v8f;

union Frag16 {
    v16bf    v;
    uint16_t u[16];
    uint4    q[2];
};

__device__ __forceinline__ uint16_t f32_bf16(float f) {
    union { float f; uint32_t u; } c;
    c.f = f;
    uint32_t u = c.u;
    u += 0x7FFFu + ((u >> 16) & 1u);   // round-to-nearest-even
    return (uint16_t)(u >> 16);
}

// Async copy of one 16B chunk: global -> LDS (tracked by ASYNCcnt).
__device__ __forceinline__ void async_g2l_16B(uint32_t lds_byte_addr,
                                              const uint16_t* gaddr) {
    asm volatile("global_load_async_to_lds_b128 %0, %1, off"
                 :: "v"(lds_byte_addr), "v"(gaddr)
                 : "memory");
}

__device__ __forceinline__ void wait_async0() {
    asm volatile("s_wait_asynccnt 0x0" ::: "memory");
}

// Issue async copies for a 128x64 bf16 tile (rows rowBase.., cols k0..k0+63,
// row stride `stride` elems) into the padded LDS tile. No wait, no barrier.
__device__ __forceinline__ void stage_issue(uint16_t* ldsTile,
                                            const uint16_t* __restrict__ src,
                                            int rowBase, int k0, int stride) {
    const uint32_t ldsBase = (uint32_t)(uintptr_t)ldsTile;
    const int row = threadIdx.x >> 1;          // 0..127
    const int subBase = (threadIdx.x & 1) * 4; // 16B-chunk base (0 or 4)
    const uint16_t* gp = src + (size_t)(rowBase + row) * stride + k0 + subBase * 8;
    uint32_t lp = ldsBase + row * (BT_ROWPAD * 2) + subBase * 16;
#pragma unroll
    for (int c = 0; c < 4; c++) {
        async_g2l_16B(lp + c * 16, gp + c * 8);
    }
}

__device__ __forceinline__ void load_bfrag(Frag16& f, const uint16_t* tile,
                                           int nt, int ks, int lr, int hi) {
    const uint16_t* bp = &tile[(nt * 16 + lr) * BT_ROWPAD + ks + hi * 8];
    f.q[0] = *(const uint4*)(bp);
    f.q[1] = *(const uint4*)(bp + 16);
}

// ---------------------------------------------------------------------------
// Pass 0: f32 [L][C] -> bf16 row-major copy AND bf16 transposed copy [C][L]
// ---------------------------------------------------------------------------
__global__ void __launch_bounds__(256)
cvt_tr_kernel(const float* __restrict__ X, uint16_t* __restrict__ Xbf,
              uint16_t* __restrict__ XT, int R, int C) {
    __shared__ float tile[32][33];
    const int bb = blockIdx.z;
    const float* Xb   = X   + (size_t)bb * R * C;
    uint16_t*    Xbfb = Xbf + (size_t)bb * R * C;
    uint16_t*    XTb  = XT  + (size_t)bb * C * R;
    const int c0 = blockIdx.x * 32, r0 = blockIdx.y * 32;
    const int tx = threadIdx.x, ty = threadIdx.y;
#pragma unroll
    for (int k = 0; k < 4; k++) {
        int r = ty * 4 + k;
        float v = __builtin_nontemporal_load(&Xb[(size_t)(r0 + r) * C + c0 + tx]);
        tile[r][tx] = v;
        Xbfb[(size_t)(r0 + r) * C + c0 + tx] = f32_bf16(v);
    }
    __syncthreads();
#pragma unroll
    for (int k = 0; k < 4; k++) {
        int cc = ty * 4 + k;
        XTb[(size_t)(c0 + cc) * R + r0 + tx] = f32_bf16(tile[tx][cc]);
    }
}

// ---------------------------------------------------------------------------
// Pass 1: e[b,i,j] = sum_d a[b,i,d]*b[b,j,d].
// Block 256 thr / 8 waves, 128x128 tile. B-tile double-buffered in LDS via
// async DMA overlapped with WMMA; all 8 B-fragments loaded ahead of the
// 8 WMMAs, enforced with a sched_barrier so the waits stagger.
// ---------------------------------------------------------------------------
__global__ void __launch_bounds__(256)
gemm_e_kernel(const uint16_t* __restrict__ Abf, const uint16_t* __restrict__ Bbf,
              float* __restrict__ E) {
    __shared__ uint16_t Bt[2][BT_ELEMS];
    const int bb   = blockIdx.z;
    const int wave = threadIdx.x >> 5;
    const int lane = threadIdx.x & 31;
    const int hi   = lane >> 4;
    const int lr   = lane & 15;
    const int i0 = blockIdx.y * 128 + wave * 16;
    const int j0 = blockIdx.x * 128;
    const uint16_t* Ab = Abf + (size_t)bb * LA * DD;
    const uint16_t* Bb = Bbf + (size_t)bb * LB * DD;

    v8f zero = {0.f, 0.f, 0.f, 0.f, 0.f, 0.f, 0.f, 0.f};
    v8f acc[8];
#pragma unroll
    for (int t = 0; t < 8; t++) acc[t] = zero;

    stage_issue(Bt[0], Bb, j0, 0, DD);
    const int NT = DD / 64;
    for (int t = 0; t < NT; t++) {
        const int k0 = t * 64;
        wait_async0();
        __syncthreads();
        if (t + 1 < NT) stage_issue(Bt[(t + 1) & 1], Bb, j0, k0 + 64, DD);
        const uint16_t* cur = Bt[t & 1];
#pragma unroll
        for (int ks = 0; ks < 64; ks += 32) {
            Frag16 af;
            const uint16_t* ap = Ab + (size_t)(i0 + lr) * DD + k0 + ks + hi * 8;
            af.q[0] = *(const uint4*)(ap);
            af.q[1] = *(const uint4*)(ap + 16);
            Frag16 bfr[8];
#pragma unroll
            for (int nt = 0; nt < 8; nt++) load_bfrag(bfr[nt], cur, nt, ks, lr, hi);
            __builtin_amdgcn_sched_barrier(0);   // keep loads clustered ahead
#pragma unroll
            for (int nt = 0; nt < 8; nt++) {
                acc[nt] = __builtin_amdgcn_wmma_f32_16x16x32_bf16(
                    false, af.v, false, bfr[nt].v, (short)0, acc[nt], false, false);
            }
        }
    }

    float* Eb = E + (size_t)bb * LA * LB;
#pragma unroll
    for (int nt = 0; nt < 8; nt++) {
#pragma unroll
        for (int r = 0; r < 8; r++) {
            Eb[(size_t)(i0 + hi * 8 + r) * LB + j0 + nt * 16 + lr] = acc[nt][r];
        }
    }
}

// ---------------------------------------------------------------------------
// Pass 2a: row stats (softmax over j). One wave per row of e.
// ---------------------------------------------------------------------------
__global__ void __launch_bounds__(256)
row_stats_kernel(const float* __restrict__ E, float* __restrict__ rmax,
                 float* __restrict__ rinv) {
    const int wave = threadIdx.x >> 5;
    const int lane = threadIdx.x & 31;
    const int row  = blockIdx.x * 8 + wave;
    const float* Er = E + (size_t)row * LB;
    float x[16];
    float m = -3.4e38f;
#pragma unroll
    for (int t = 0; t < 16; t++) {
        x[t] = Er[lane + t * 32];
        m = fmaxf(m, x[t]);
    }
#pragma unroll
    for (int off = 16; off > 0; off >>= 1) m = fmaxf(m, __shfl_xor(m, off, 32));
    float s = 0.f;
#pragma unroll
    for (int t = 0; t < 16; t++) s += __expf(x[t] - m);
#pragma unroll
    for (int off = 16; off > 0; off >>= 1) s += __shfl_xor(s, off, 32);
    if (lane == 0) {
        rmax[row] = m;
        rinv[row] = 1.f / s;
    }
}

// ---------------------------------------------------------------------------
// Pass 2b: column stats (softmax over i). One thread per column, online.
// ---------------------------------------------------------------------------
__global__ void __launch_bounds__(256)
col_stats_kernel(const float* __restrict__ E, float* __restrict__ cmax,
                 float* __restrict__ cinv) {
    const int bb = blockIdx.y;
    const int j  = blockIdx.x * 256 + threadIdx.x;
    const float* Eb = E + (size_t)bb * LA * LB;
    float m = -3.4e38f, s = 0.f;
    for (int i = 0; i < LA; i++) {
        float xv = Eb[(size_t)i * LB + j];
        float nm = fmaxf(m, xv);
        s = s * __expf(m - nm) + __expf(xv - nm);
        m = nm;
    }
    cmax[bb * LB + j] = m;
    cinv[bb * LB + j] = 1.f / s;
}

// ---------------------------------------------------------------------------
// Pass 3: a_tilda = softmax_row(e) @ b.   A=P built in regs, B=bT via LDS.
// ---------------------------------------------------------------------------
__global__ void __launch_bounds__(256)
gemm_at_kernel(const float* __restrict__ E, const uint16_t* __restrict__ BT,
               const float* __restrict__ rmax, const float* __restrict__ rinv,
               float* __restrict__ out) {
    __shared__ uint16_t Bt[2][BT_ELEMS];
    const int bb   = blockIdx.z;
    const int wave = threadIdx.x >> 5;
    const int lane = threadIdx.x & 31;
    const int hi   = lane >> 4;
    const int lr   = lane & 15;
    const int i0 = blockIdx.y * 128 + wave * 16;
    const int d0 = blockIdx.x * 128;
    const float*    Eb  = E  + (size_t)bb * LA * LB;
    const uint16_t* BTb = BT + (size_t)bb * DD * LB;   // [D][LB]
    const int row = i0 + lr;
    const float rm = rmax[bb * LA + row];
    const float ri = rinv[bb * LA + row];

    v8f zero = {0.f, 0.f, 0.f, 0.f, 0.f, 0.f, 0.f, 0.f};
    v8f acc[8];
#pragma unroll
    for (int t = 0; t < 8; t++) acc[t] = zero;

    stage_issue(Bt[0], BTb, d0, 0, LB);
    const int NT = LB / 64;
    for (int t = 0; t < NT; t++) {
        const int k0 = t * 64;
        wait_async0();
        __syncthreads();
        if (t + 1 < NT) stage_issue(Bt[(t + 1) & 1], BTb, d0, k0 + 64, LB);
        const uint16_t* cur = Bt[t & 1];
#pragma unroll
        for (int ks = 0; ks < 64; ks += 32) {
            const int kb = k0 + ks + hi * 8;
            const float* ep = Eb + (size_t)row * LB + kb;
            Frag16 af;
#pragma unroll
            for (int t2 = 0; t2 < 8; t2++)
                af.u[t2] = f32_bf16(__expf(ep[t2] - rm) * ri);
#pragma unroll
            for (int t2 = 0; t2 < 8; t2++)
                af.u[8 + t2] = f32_bf16(__expf(ep[16 + t2] - rm) * ri);
            Frag16 bfr[8];
#pragma unroll
            for (int nt = 0; nt < 8; nt++) load_bfrag(bfr[nt], cur, nt, ks, lr, hi);
            __builtin_amdgcn_sched_barrier(0);   // keep loads clustered ahead
#pragma unroll
            for (int nt = 0; nt < 8; nt++) {
                acc[nt] = __builtin_amdgcn_wmma_f32_16x16x32_bf16(
                    false, af.v, false, bfr[nt].v, (short)0, acc[nt], false, false);
            }
        }
    }

    float* ob = out + (size_t)bb * LA * DD;
#pragma unroll
    for (int nt = 0; nt < 8; nt++) {
#pragma unroll
        for (int r = 0; r < 8; r++) {
            __builtin_nontemporal_store(
                acc[nt][r],
                &ob[(size_t)(i0 + hi * 8 + r) * DD + d0 + nt * 16 + lr]);
        }
    }
}

// ---------------------------------------------------------------------------
// Pass 4: b_tilda = softmax_col(e)^T @ a.  A=P^T built in regs, B=aT via LDS.
// ---------------------------------------------------------------------------
__global__ void __launch_bounds__(256)
gemm_bt_kernel(const float* __restrict__ E, const uint16_t* __restrict__ AT,
               const float* __restrict__ cmax, const float* __restrict__ cinv,
               float* __restrict__ out) {
    __shared__ uint16_t Bt[2][BT_ELEMS];
    const int bb   = blockIdx.z;
    const int wave = threadIdx.x >> 5;
    const int lane = threadIdx.x & 31;
    const int hi   = lane >> 4;
    const int lr   = lane & 15;
    const int j0 = blockIdx.y * 128 + wave * 16;
    const int d0 = blockIdx.x * 128;
    const float*    Eb  = E  + (size_t)bb * LA * LB;
    const uint16_t* ATb = AT + (size_t)bb * DD * LA;   // [D][LA]
    const int col = j0 + lr;
    const float cm = cmax[bb * LB + col];
    const float ci = cinv[bb * LB + col];

    v8f zero = {0.f, 0.f, 0.f, 0.f, 0.f, 0.f, 0.f, 0.f};
    v8f acc[8];
#pragma unroll
    for (int t = 0; t < 8; t++) acc[t] = zero;

    stage_issue(Bt[0], ATb, d0, 0, LA);
    const int NT = LA / 64;
    for (int t = 0; t < NT; t++) {
        const int k0 = t * 64;
        wait_async0();
        __syncthreads();
        if (t + 1 < NT) stage_issue(Bt[(t + 1) & 1], ATb, d0, k0 + 64, LA);
        const uint16_t* cur = Bt[t & 1];
#pragma unroll
        for (int ks = 0; ks < 64; ks += 32) {
            const int kb = k0 + ks + hi * 8;      // K = i index
            Frag16 af;
#pragma unroll
            for (int t2 = 0; t2 < 8; t2++)
                af.u[t2] =
                    f32_bf16(__expf(Eb[(size_t)(kb + t2) * LB + col] - cm) * ci);
#pragma unroll
            for (int t2 = 0; t2 < 8; t2++)
                af.u[8 + t2] =
                    f32_bf16(__expf(Eb[(size_t)(kb + 16 + t2) * LB + col] - cm) * ci);
            Frag16 bfr[8];
#pragma unroll
            for (int nt = 0; nt < 8; nt++) load_bfrag(bfr[nt], cur, nt, ks, lr, hi);
            __builtin_amdgcn_sched_barrier(0);   // keep loads clustered ahead
#pragma unroll
            for (int nt = 0; nt < 8; nt++) {
                acc[nt] = __builtin_amdgcn_wmma_f32_16x16x32_bf16(
                    false, af.v, false, bfr[nt].v, (short)0, acc[nt], false, false);
            }
        }
    }

    float* ob = out + (size_t)bb * LB * DD;
#pragma unroll
    for (int nt = 0; nt < 8; nt++) {
#pragma unroll
        for (int r = 0; r < 8; r++) {
            __builtin_nontemporal_store(
                acc[nt][r],
                &ob[(size_t)(j0 + hi * 8 + r) * DD + d0 + nt * 16 + lr]);
        }
    }
}

// ---------------------------------------------------------------------------
extern "C" void kernel_launch(void* const* d_in, const int* in_sizes, int n_in,
                              void* d_out, int out_size, void* d_ws, size_t ws_size,
                              hipStream_t stream) {
    const float* a = (const float*)d_in[0];   // [B, LA, D] f32
    const float* b = (const float*)d_in[1];   // [B, LB, D] f32

    const size_t nAD = (size_t)B_ * LA * DD;
    const size_t nBD = (size_t)B_ * LB * DD;
    const size_t nE  = (size_t)B_ * LA * LB;

    char* ws = (char*)d_ws;
    size_t off = 0;
    uint16_t* a_bf  = (uint16_t*)(ws + off); off += nAD * 2;
    uint16_t* b_bf  = (uint16_t*)(ws + off); off += nBD * 2;
    uint16_t* aT_bf = (uint16_t*)(ws + off); off += nAD * 2;
    uint16_t* bT_bf = (uint16_t*)(ws + off); off += nBD * 2;
    float*    e     = (float*)(ws + off);    off += nE * 4;
    float*    rmax  = (float*)(ws + off);    off += (size_t)B_ * LA * 4;
    float*    rinv  = (float*)(ws + off);    off += (size_t)B_ * LA * 4;
    float*    cmaxp = (float*)(ws + off);    off += (size_t)B_ * LB * 4;
    float*    cinvp = (float*)(ws + off);    off += (size_t)B_ * LB * 4;

    float* out_a = (float*)d_out;
    float* out_b = out_a + nAD;

    cvt_tr_kernel<<<dim3(DD / 32, LA / 32, B_), dim3(32, 8), 0, stream>>>(
        a, a_bf, aT_bf, LA, DD);
    cvt_tr_kernel<<<dim3(DD / 32, LB / 32, B_), dim3(32, 8), 0, stream>>>(
        b, b_bf, bT_bf, LB, DD);

    gemm_e_kernel<<<dim3(LB / 128, LA / 128, B_), 256, 0, stream>>>(a_bf, b_bf, e);

    row_stats_kernel<<<dim3((B_ * LA) / 8), 256, 0, stream>>>(e, rmax, rinv);
    col_stats_kernel<<<dim3(LB / 256, B_), 256, 0, stream>>>(e, cmaxp, cinvp);

    gemm_at_kernel<<<dim3(DD / 128, LA / 128, B_), 256, 0, stream>>>(
        e, bT_bf, rmax, rinv, out_a);
    gemm_bt_kernel<<<dim3(DD / 128, LB / 128, B_), 256, 0, stream>>>(
        e, aT_bf, cmaxp, cinvp, out_b);
}
